// Stack_NMN_10806137716760
// MI455X (gfx1250) — matrix operand order repaired and checked by reading, not compiled, not run on hardware
//
#include <hip/hip_runtime.h>
#include <hip/hip_bf16.h>
#include <math.h>

typedef __attribute__((ext_vector_type(16))) _Float16 v16h;
typedef __attribute__((ext_vector_type(8)))  float    v8f;

#define B_    64
#define S_    30
#define E_    300
#define HL_   256
#define TWOH_ 512
#define KB_   512
#define MAPD_ 512
#define MEMD_ 512
#define MLP_  1024
#define NCLS_ 32
#define L_    8
#define T_    9
#define NMOD_ 9
#define HW_   150
#define CIN_  640
#define NPIX_ (B_*HW_)   /* 9600 */

__device__ __forceinline__ float eluf(float x){ return x > 0.f ? x : (expf(x) - 1.f); }
__device__ __forceinline__ float sigmoidf_(float x){ return 1.f / (1.f + expf(-x)); }

// ---------------------------------------------------------------------------
// WMMA GEMM:  D = act( A(MxK,f32) * B(KxN,f32, strided) + bias[n] + resid )
// Template parameters (resolved at launch -> branch-free hot loop):
//   BVEC : B is K-contiguous (sbk==1, W[o][c] weights) -> 4x float4 loads.
//          Otherwise scalar loads, coalesced across lanes (sbn==1 layouts).
//   KTAIL: compile the guarded K%32 remainder chunk (only K=300 GEMMs).
//   FULLN: N % 64 == 0 -> constant 4 subtiles per wave, no break test.
// Requirements at every call site: M%16==0, N%16==0, lda%4==0, 16B-aligned A.
// 8 waves / block, wave = 16 rows x 64 cols (A-frag reused over 4 B-frags).
// ---------------------------------------------------------------------------
template<bool BVEC, bool KTAIL, bool FULLN>
__global__ __launch_bounds__(256)
void wmma_gemm_t(const float* __restrict__ A, int lda,
                 const float* __restrict__ Bm, int sbk, int sbn,
                 const float* __restrict__ bias,
                 const float* __restrict__ resid, int ldr,
                 float* __restrict__ D, int ldd,
                 int M, int N, int K, int act)
{
  const int wave = threadIdx.x >> 5;
  const int lane = threadIdx.x & 31;
  const int m0 = blockIdx.x * 128 + wave * 16;
  const int n0 = blockIdx.y * 64;
  if (m0 >= M) return;                        // uniform per wave (M % 16 == 0)
  const int half = lane >> 4;
  const int mr   = lane & 15;
  const float* Arow = A + (size_t)(m0 + mr) * lda;

  int nsub;
  if (FULLN) nsub = 4;
  else { nsub = (N - n0) >> 4; if (nsub > 4) nsub = 4; }

  v8f acc[4];
  #pragma unroll
  for (int t = 0; t < 4; ++t)
    #pragma unroll
    for (int i = 0; i < 8; ++i) acc[t][i] = 0.f;

  const int kmain = K & ~31;
  for (int kb = 0; kb < kmain; kb += 32) {
    // ---- A fragment (K = kb + half*16 + j) ----
    const float* a0p = Arow + kb + half * 16;
    float4 x0 = *(const float4*)(a0p);
    float4 x1 = *(const float4*)(a0p + 4);
    float4 x2 = *(const float4*)(a0p + 8);
    float4 x3 = *(const float4*)(a0p + 12);
    __builtin_prefetch(a0p + 64, 0, 1);       // speculative; OOB is dropped
    v16h afrag;
    afrag[0]=(_Float16)x0.x;  afrag[1]=(_Float16)x0.y;
    afrag[2]=(_Float16)x0.z;  afrag[3]=(_Float16)x0.w;
    afrag[4]=(_Float16)x1.x;  afrag[5]=(_Float16)x1.y;
    afrag[6]=(_Float16)x1.z;  afrag[7]=(_Float16)x1.w;
    afrag[8]=(_Float16)x2.x;  afrag[9]=(_Float16)x2.y;
    afrag[10]=(_Float16)x2.z; afrag[11]=(_Float16)x2.w;
    afrag[12]=(_Float16)x3.x; afrag[13]=(_Float16)x3.y;
    afrag[14]=(_Float16)x3.z; afrag[15]=(_Float16)x3.w;

    // ---- B fragments + WMMA ----
    #pragma unroll
    for (int t = 0; t < 4; ++t) {
      if (!FULLN && t >= nsub) break;
      int n = n0 + t * 16 + mr;
      v16h bfrag;
      if (BVEC) {
        const float* bp = Bm + (size_t)(kb + half * 16) + (size_t)n * sbn;
        float4 b0 = *(const float4*)(bp);
        float4 b1 = *(const float4*)(bp + 4);
        float4 b2 = *(const float4*)(bp + 8);
        float4 b3 = *(const float4*)(bp + 12);
        bfrag[0]=(_Float16)b0.x;  bfrag[1]=(_Float16)b0.y;
        bfrag[2]=(_Float16)b0.z;  bfrag[3]=(_Float16)b0.w;
        bfrag[4]=(_Float16)b1.x;  bfrag[5]=(_Float16)b1.y;
        bfrag[6]=(_Float16)b1.z;  bfrag[7]=(_Float16)b1.w;
        bfrag[8]=(_Float16)b2.x;  bfrag[9]=(_Float16)b2.y;
        bfrag[10]=(_Float16)b2.z; bfrag[11]=(_Float16)b2.w;
        bfrag[12]=(_Float16)b3.x; bfrag[13]=(_Float16)b3.y;
        bfrag[14]=(_Float16)b3.z; bfrag[15]=(_Float16)b3.w;
      } else {
        const float* bp = Bm + (size_t)(kb + half * 16) * sbk + (size_t)n * sbn;
        #pragma unroll
        for (int j = 0; j < 16; ++j)
          bfrag[j] = (_Float16)bp[(size_t)j * sbk];
      }
      acc[t] = __builtin_amdgcn_wmma_f32_16x16x32_f16(false, afrag, false, bfrag,
                                                      (short)0, acc[t], false, false);
    }
  }

  // ---- guarded K tail (compiled only for K % 32 != 0, e.g. K = 300) ----
  if (KTAIL) {
    v16h afrag;
    #pragma unroll
    for (int j = 0; j < 16; ++j) {
      int kk = kmain + half * 16 + j;
      afrag[j] = (kk < K) ? (_Float16)Arow[kk] : (_Float16)0.f;
    }
    #pragma unroll
    for (int t = 0; t < 4; ++t) {
      if (!FULLN && t >= nsub) break;
      int n = n0 + t * 16 + mr;
      v16h bfrag;
      #pragma unroll
      for (int j = 0; j < 16; ++j) {
        int kk = kmain + half * 16 + j;
        bfrag[j] = (kk < K) ? (_Float16)Bm[(size_t)kk * sbk + (size_t)n * sbn]
                            : (_Float16)0.f;
      }
      acc[t] = __builtin_amdgcn_wmma_f32_16x16x32_f16(false, afrag, false, bfrag,
                                                      (short)0, acc[t], false, false);
    }
  }

  // ---- epilogue ----
  #pragma unroll
  for (int t = 0; t < 4; ++t) {
    if (!FULLN && t >= nsub) break;
    int n = n0 + t * 16 + mr;
    #pragma unroll
    for (int i = 0; i < 8; ++i) {             // c[i] = C[M = i + 8*half][N = lane%16]
      int r = m0 + half * 8 + i;
      float v = acc[t][i];
      if (bias)  v += bias[n];
      if (resid) v += resid[(size_t)r * ldr + n];
      if (act == 1) v = eluf(v);
      D[(size_t)r * ldd + n] = v;
    }
  }
}

// ---------------------------------------------------------------------------
// Small helper kernels
// ---------------------------------------------------------------------------
__global__ void k_zero(float* p, int n){
  int i = blockIdx.x*blockDim.x + threadIdx.x;
  if (i < n) p[i] = 0.f;
}

__global__ void k_pinit(float* p){           // p[:,0]=1, rest 0 (64x8)
  int i = blockIdx.x*blockDim.x + threadIdx.x;
  if (i < B_*L_) p[i] = ((i & 7) == 0) ? 1.f : 0.f;
}

__global__ void k_im2row(const float* __restrict__ img, float* __restrict__ out){
  int i = blockIdx.x*blockDim.x + threadIdx.x;
  if (i >= NPIX_*CIN_) return;
  int r = i / CIN_, c = i % CIN_;
  int b = r / HW_,  hw = r % HW_;
  out[i] = img[((size_t)b*CIN_ + c)*HW_ + hw];
}

__global__ void k_embed(const int* __restrict__ ques, const float* __restrict__ W,
                        float* __restrict__ emb){
  int i = blockIdx.x*blockDim.x + threadIdx.x;
  if (i >= B_*S_*E_) return;
  int r = i / E_, e = i % E_;
  emb[i] = W[(size_t)ques[r]*E_ + e];
}

__global__ void k_lstm_gates(const float* __restrict__ z, const int* __restrict__ q_len,
                             int s, float* __restrict__ h, int ldh, float* __restrict__ c,
                             float* __restrict__ lstm_out, int outOff){
  int i = blockIdx.x*blockDim.x + threadIdx.x;
  if (i >= B_*HL_) return;
  int b = i >> 8, n = i & 255;
  const float* zb = z + b*1024;
  float ig = sigmoidf_(zb[n]);
  float fg = sigmoidf_(zb[HL_   + n]);
  float gg = tanhf   (zb[2*HL_ + n]);
  float og = sigmoidf_(zb[3*HL_ + n]);
  float cp = c[i];
  float cn = fg*cp + ig*gg;
  float hn = og*tanhf(cn);
  bool m = s < q_len[b];
  if (m) { h[b*ldh + n] = hn; c[i] = cn; }
  lstm_out[((size_t)b*S_ + s)*TWOH_ + outOff + n] = m ? hn : 0.f;
}

__global__ void k_tile_c0(const float* __restrict__ c0, float* __restrict__ ucat){
  int i = blockIdx.x*blockDim.x + threadIdx.x;
  if (i >= B_*TWOH_) return;
  int b = i >> 9, n = i & 511;
  ucat[b*1024 + 512 + n] = c0[n];
}

__global__ void k_mixw(const float* __restrict__ c3ow, const float* __restrict__ c4ow,
                       const float* __restrict__ aow, float* __restrict__ mixw){
  int j = threadIdx.x;
  if (j >= 8) return;
  const float* src = (j < 4) ? c3ow + j*6 : (j < 6) ? c4ow + (j-4)*6 : aow + (j-6)*6;
  float mx = src[0];
  for (int k = 1; k < 6; ++k) mx = fmaxf(mx, src[k]);
  float e[6]; float se = 0.f;
  for (int k = 0; k < 6; ++k){ e[k] = expf(src[k]-mx); se += e[k]; }
  for (int k = 0; k < 6; ++k) mixw[j*6+k] = e[k]/se;
}

// wl = u @ Wmod ; log_softmax ; w = exp(ls) ; ent[b] -= sum w*ls
__global__ void k_wl(const float* __restrict__ u, const float* __restrict__ Wmod,
                     float* __restrict__ wprob, float* __restrict__ ent){
  int b = blockIdx.x, tid = threadIdx.x;
  __shared__ float wl[NMOD_];
  if (tid < NMOD_) {
    const float* ub = u + b*TWOH_;
    float s = 0.f;
    for (int k = 0; k < TWOH_; ++k) s += ub[k]*Wmod[k*NMOD_ + tid];
    wl[tid] = s;
  }
  __syncthreads();
  if (tid == 0){
    float mx = wl[0];
    for (int j = 1; j < NMOD_; ++j) mx = fmaxf(mx, wl[j]);
    float se = 0.f;
    for (int j = 0; j < NMOD_; ++j) se += expf(wl[j]-mx);
    float lse = logf(se) + mx;
    float e = 0.f;
    for (int j = 0; j < NMOD_; ++j){
      float ls = wl[j] - lse;
      float w  = expf(ls);
      wprob[b*NMOD_ + j] = w;
      e -= w*ls;
    }
    ent[b] += e;
  }
}

// attn over S, writes c -> ucat[:,512:1024]
__global__ void k_attn(const float* __restrict__ lstm_out, const float* __restrict__ u,
                       const float* __restrict__ W3, float* __restrict__ ucat){
  int b = blockIdx.x, tid = threadIdx.x;
  __shared__ float aw[S_];
  if (tid < S_) {
    const float* lo = lstm_out + ((size_t)b*S_ + tid)*TWOH_;
    const float* ub = u + b*TWOH_;
    float s = 0.f;
    for (int h = 0; h < TWOH_; ++h) s += lo[h]*ub[h]*W3[h];
    aw[tid] = s;
  }
  __syncthreads();
  if (tid == 0){
    float mx = aw[0];
    for (int s = 1; s < S_; ++s) mx = fmaxf(mx, aw[s]);
    float se = 0.f;
    for (int s = 0; s < S_; ++s){ aw[s] = expf(aw[s]-mx); se += aw[s]; }
    for (int s = 0; s < S_; ++s) aw[s] /= se;
  }
  __syncthreads();
  for (int n = tid; n < TWOH_; n += blockDim.x){
    float cv = 0.f;
    for (int s = 0; s < S_; ++s) cv += aw[s]*lstm_out[((size_t)b*S_ + s)*TWOH_ + n];
    ucat[b*1024 + 512 + n] = cv;
  }
}

__global__ void k_atop(const float* __restrict__ stack, const float* __restrict__ p,
                       float* __restrict__ atop, float* __restrict__ a2){
  int i = blockIdx.x*blockDim.x + threadIdx.x;
  if (i >= NPIX_) return;
  int b = i / HW_;
  const float* pb = p + b*L_;
  float s1 = 0.f, s2 = 0.f;
  #pragma unroll
  for (int l = 0; l < L_; ++l){
    float pd = ((l < 7) ? pb[l+1] : 0.f) + ((l == 0) ? pb[0] : 0.f);
    float sv = stack[(size_t)i*L_ + l];
    s1 += sv*pb[l]; s2 += sv*pd;
  }
  atop[i] = s1; a2[i] = s2;
}

// 6 cell outputs per pixel (4x mix3, 2x mix4) fused: one pass over d
__global__ __launch_bounds__(256)
void k_cellout(const float* __restrict__ imgmap, const float* __restrict__ tmap,
               const float* __restrict__ atop, const float* __restrict__ a2,
               const float* __restrict__ mixw,
               const float* __restrict__ c3cw, const float* __restrict__ c3cb,
               const float* __restrict__ c4cw, const float* __restrict__ c4cb,
               float* __restrict__ cellout){
  int pix = blockIdx.x;
  int b   = pix / HW_;
  int tid = threadIdx.x;
  float at  = atop[pix];
  float av2 = a2[pix];
  float acc[6];
  #pragma unroll
  for (int m = 0; m < 6; ++m) acc[m] = 0.f;
  const float* xr = imgmap + (size_t)pix*MAPD_;
  const float* tr = tmap + b*MAPD_;
  for (int d = tid; d < MAPD_; d += 256){
    float x = xr[d], tv = tr[d];
    #pragma unroll
    for (int m = 0; m < 4; ++m){
      const float* w = mixw + m*6;
      float val = x*(w[0]*tv + w[1] + w[4] + at*(w[2] + w[3]*tv)) + tv*(w[1] + w[5]*at);
      acc[m] += eluf(val)*c3cw[m*MAPD_ + d];
    }
    #pragma unroll
    for (int m = 0; m < 2; ++m){
      const float* w = mixw + (4+m)*6;
      float val = x*(w[0]*tv + w[1] + w[2]*at + w[3]*av2 + w[4]*at*av2 + w[5]*tv*(at+av2))
                + w[1]*tv;
      acc[4+m] += eluf(val)*c4cw[m*MAPD_ + d];
    }
  }
  __shared__ float red[256];
  for (int m = 0; m < 6; ++m){
    red[tid] = acc[m]; __syncthreads();
    for (int off = 128; off > 0; off >>= 1){
      if (tid < off) red[tid] += red[tid+off];
      __syncthreads();
    }
    if (tid == 0){
      float bias = (m < 4) ? c3cb[m] : c4cb[m-4];
      cellout[(size_t)m*NPIX_ + pix] = red[0] + bias;
    }
    __syncthreads();
  }
}

// pooled answer mixes (mean over HW) + tmap copy into concat buffers
__global__ void k_ansmix(const float* __restrict__ imgmap, const float* __restrict__ tmap,
                         const float* __restrict__ atop, const float* __restrict__ a2,
                         const float* __restrict__ mixw,
                         float* __restrict__ mcat6, float* __restrict__ mcat7){
  int i = blockIdx.x*blockDim.x + threadIdx.x;
  if (i >= B_*MAPD_) return;
  int b = i >> 9, d = i & 511;
  float tv = tmap[i];
  const float* w6 = mixw + 6*6;
  const float* w7 = mixw + 7*6;
  float s6 = 0.f, s7 = 0.f;
  for (int hw = 0; hw < HW_; ++hw){
    int pix = b*HW_ + hw;
    float x   = imgmap[(size_t)pix*MAPD_ + d];
    float at  = atop[pix];
    float av2 = a2[pix];
    float v6 = x*(w6[0]*tv + w6[1] + w6[4] + at*(w6[2] + w6[3]*tv)) + tv*(w6[1] + w6[5]*at);
    s6 += eluf(v6);
    float v7 = x*(w7[0]*tv + w7[1] + w7[2]*at + w7[3]*av2 + w7[4]*at*av2 + w7[5]*tv*(at+av2))
             + w7[1]*tv;
    s7 += eluf(v7);
  }
  mcat6[b*1024 + d] = s6 * (1.f/HW_);
  mcat7[b*1024 + d] = s7 * (1.f/HW_);
  mcat6[b*1024 + 512 + d] = tv;
  mcat7[b*1024 + 512 + d] = tv;
}

// stack <- sum_m w_m * stacks_m  (in place; element-wise)
__global__ void k_stack_update(float* __restrict__ stack, const float* __restrict__ p,
                               const float* __restrict__ w, const float* __restrict__ cellout){
  int i = blockIdx.x*blockDim.x + threadIdx.x;
  if (i >= NPIX_) return;
  int b = i / HW_;
  const float* wb = w + b*NMOD_;
  float wA = wb[0]+wb[1]+wb[2]+wb[3];
  float wB = wb[4]+wb[5];
  float o3 = wb[0]*cellout[i] + wb[1]*cellout[NPIX_+i]
           + wb[2]*cellout[2*NPIX_+i] + wb[3]*cellout[3*NPIX_+i];
  float o4 = wb[4]*cellout[4*NPIX_+i] + wb[5]*cellout[5*NPIX_+i];
  const float* pb = p + b*L_;
  #pragma unroll
  for (int l = 0; l < L_; ++l){
    float pv = pb[l];
    float pd = ((l < 7) ? pb[l+1] : 0.f) + ((l == 0) ? pb[0] : 0.f);
    size_t si = (size_t)i*L_ + l;
    stack[si] = stack[si]*(1.f - wA*pv - wB*pd) + o3*pv + o4*pd;
  }
}

__global__ void k_p_update(const float* __restrict__ w, const float* __restrict__ pin,
                           float* __restrict__ pout){
  int i = blockIdx.x*blockDim.x + threadIdx.x;
  if (i >= B_*L_) return;
  int b = i >> 3, l = i & 7;
  const float* wb = w + b*NMOD_;
  float alpha = wb[0]+wb[1]+wb[2]+wb[3]+wb[6]+wb[8];
  float beta  = wb[4]+wb[5]+wb[7];
  float pv = pin[i];
  float pd = ((l < 7) ? pin[b*L_+l+1] : 0.f) + ((l == 0) ? pin[b*L_] : 0.f);
  pout[i] = alpha*pv + beta*pd;
}

__global__ void k_mem_update(const float* __restrict__ w, const float* __restrict__ mem6,
                             const float* __restrict__ mem7, float* __restrict__ mem){
  int i = blockIdx.x*blockDim.x + threadIdx.x;
  if (i >= B_*MEMD_) return;
  int b = i >> 9;
  float w6 = w[b*NMOD_ + 6], w7 = w[b*NMOD_ + 7];
  mem[i] = w6*mem6[i] + w7*mem7[i] + (1.f - w6 - w7)*mem[i];
}

__global__ void k_qm(const float* __restrict__ q, const float* __restrict__ mem,
                     float* __restrict__ qm){
  int i = blockIdx.x*blockDim.x + threadIdx.x;
  if (i >= B_*1024) return;
  int b = i >> 10, n = i & 1023;
  qm[i] = (n < 512) ? q[b*512 + n] : mem[b*512 + (n-512)];
}

__global__ void k_entred(const float* __restrict__ ent, float* __restrict__ out){
  if (blockIdx.x == 0 && threadIdx.x == 0){
    float s = 0.f;
    for (int i = 0; i < B_; ++i) s += ent[i];   // fixed order -> deterministic
    out[0] = s;
  }
}

// ---------------------------------------------------------------------------
// Dispatcher: picks the specialized instantiation once per launch.
// ---------------------------------------------------------------------------
static inline void gemm(hipStream_t st, const float* A, int lda,
                        const float* Bm, int sbk, int sbn,
                        const float* bias, const float* resid, int ldr,
                        float* D, int ldd, int M, int N, int K, int act){
  dim3 g((M + 127)/128, (N + 63)/64);
  const bool bvec  = (sbk == 1) && ((sbn & 3) == 0);
  const bool ktail = (K & 31) != 0;
  const bool fulln = (N & 63) == 0;
  #define GEMM_CALL(BV,KT,FN) \
    wmma_gemm_t<BV,KT,FN><<<g, 256, 0, st>>>(A, lda, Bm, sbk, sbn, bias, resid, ldr, \
                                             D, ldd, M, N, K, act)
  if (bvec) {
    if (ktail) { if (fulln) GEMM_CALL(true,true,true);  else GEMM_CALL(true,true,false); }
    else       { if (fulln) GEMM_CALL(true,false,true); else GEMM_CALL(true,false,false); }
  } else {
    if (ktail) { if (fulln) GEMM_CALL(false,true,true);  else GEMM_CALL(false,true,false); }
    else       { if (fulln) GEMM_CALL(false,false,true); else GEMM_CALL(false,false,false); }
  }
  #undef GEMM_CALL
}

extern "C" void kernel_launch(void* const* d_in, const int* in_sizes, int n_in,
                              void* d_out, int out_size, void* d_ws, size_t ws_size,
                              hipStream_t stream) {
  (void)in_sizes; (void)n_in; (void)out_size; (void)ws_size;
  const float* image     = (const float*)d_in[0];
  const int*   ques      = (const int*)  d_in[1];
  const int*   q_len     = (const int*)  d_in[2];
  const float* embed_W   = (const float*)d_in[3];
  const float* lstm_Wi_f = (const float*)d_in[4];
  const float* lstm_Wh_f = (const float*)d_in[5];
  const float* lstm_b_f  = (const float*)d_in[6];
  const float* lstm_Wi_b = (const float*)d_in[7];
  const float* lstm_Wh_b = (const float*)d_in[8];
  const float* lstm_b_b  = (const float*)d_in[9];
  const float* W1        = (const float*)d_in[10];
  const float* b_1       = (const float*)d_in[11];
  const float* W2_w      = (const float*)d_in[12];
  const float* W2_b      = (const float*)d_in[13];
  const float* W3        = (const float*)d_in[14];
  const float* c_0       = (const float*)d_in[15];
  const float* Wmod      = (const float*)d_in[16];
  const float* ic1_w     = (const float*)d_in[17];
  const float* ic1_b     = (const float*)d_in[18];
  const float* ic2_w     = (const float*)d_in[19];
  const float* ic2_b     = (const float*)d_in[20];
  const float* i2m_w     = (const float*)d_in[21];
  const float* i2m_b     = (const float*)d_in[22];
  const float* t2m_w     = (const float*)d_in[23];
  const float* t2m_b     = (const float*)d_in[24];
  const float* cell3_ow  = (const float*)d_in[25];
  const float* cell3_cw  = (const float*)d_in[26];
  const float* cell3_cb  = (const float*)d_in[27];
  const float* cell4_ow  = (const float*)d_in[28];
  const float* cell4_cw  = (const float*)d_in[29];
  const float* cell4_cb  = (const float*)d_in[30];
  const float* ans_ow    = (const float*)d_in[31];
  const float* ans_W     = (const float*)d_in[32];
  const float* ans_b     = (const float*)d_in[33];
  const float* out1_w    = (const float*)d_in[34];
  const float* out1_b    = (const float*)d_in[35];
  const float* out2_w    = (const float*)d_in[36];
  const float* out2_b    = (const float*)d_in[37];
  float* out = (float*)d_out;

  // ---- workspace layout (floats) ----
  float* W = (float*)d_ws;
  size_t o = 0;
  auto alloc = [&](size_t n){ float* p = W + o; o += n; return p; };
  float* x0      = alloc((size_t)NPIX_*CIN_);   // im2row; later aliased as x2
  float* x1      = alloc((size_t)NPIX_*KB_);
  float* imgmap  = alloc((size_t)NPIX_*MAPD_);
  float* emb     = alloc((size_t)B_*S_*E_);
  float* xWf     = alloc((size_t)B_*S_*1024);
  float* xWb     = alloc((size_t)B_*S_*1024);
  float* zbuf    = alloc((size_t)B_*1024);
  float* lstmout = alloc((size_t)B_*S_*TWOH_);
  float* q       = alloc((size_t)B_*TWOH_);     // [hF | hB]
  float* cstate  = alloc((size_t)2*B_*HL_);
  float* ucat    = alloc((size_t)B_*1024);      // [u1 | c_prev]
  float* u       = alloc((size_t)B_*TWOH_);
  float* tmap    = alloc((size_t)B_*MAPD_);
  float* wprob   = alloc((size_t)B_*NMOD_);
  float* ent     = alloc((size_t)B_);
  float* stack   = alloc((size_t)NPIX_*L_);
  float* pA      = alloc((size_t)B_*L_);
  float* pB      = alloc((size_t)B_*L_);
  float* atop    = alloc((size_t)NPIX_);
  float* a2b     = alloc((size_t)NPIX_);
  float* cellout = alloc((size_t)6*NPIX_);
  float* mcat6   = alloc((size_t)B_*1024);
  float* mcat7   = alloc((size_t)B_*1024);
  float* mem6    = alloc((size_t)B_*MEMD_);
  float* mem7    = alloc((size_t)B_*MEMD_);
  float* mem     = alloc((size_t)B_*MEMD_);
  float* mixw    = alloc((size_t)48);
  float* qm      = alloc((size_t)B_*1024);
  float* h1      = alloc((size_t)B_*MLP_);
  float* x2 = x0;                               // alias: x0 dead after conv1

  const int TPB = 256;
  auto nblk = [](size_t n, int t){ return (unsigned)((n + t - 1)/t); };

  // ---- init ----
  k_zero<<<nblk(B_*TWOH_,TPB),TPB,0,stream>>>(q,      B_*TWOH_);
  k_zero<<<nblk(2*B_*HL_,TPB),TPB,0,stream>>>(cstate, 2*B_*HL_);
  k_zero<<<nblk((size_t)NPIX_*L_,TPB),TPB,0,stream>>>(stack, NPIX_*L_);
  k_zero<<<nblk(B_*MEMD_,TPB),TPB,0,stream>>>(mem,    B_*MEMD_);
  k_zero<<<nblk(B_,TPB),TPB,0,stream>>>(ent,          B_);
  k_pinit<<<nblk(B_*L_,TPB),TPB,0,stream>>>(pA);

  // ---- image pipeline: im2row + 3 WMMA GEMMs ----
  k_im2row<<<nblk((size_t)NPIX_*CIN_,TPB),TPB,0,stream>>>(image, x0);
  gemm(stream, x0, CIN_, ic1_w, 1, CIN_, ic1_b, nullptr,0, x1,     KB_,   NPIX_, KB_,   CIN_, 1);
  gemm(stream, x1, KB_,  ic2_w, 1, KB_,  ic2_b, nullptr,0, x2,     KB_,   NPIX_, KB_,   KB_,  0);
  gemm(stream, x2, KB_,  i2m_w, 1, KB_,  i2m_b, nullptr,0, imgmap, MAPD_, NPIX_, MAPD_, KB_,  0);

  // ---- embedding + LSTM input projections ----
  k_embed<<<nblk((size_t)B_*S_*E_,TPB),TPB,0,stream>>>(ques, embed_W, emb);
  gemm(stream, emb, E_, lstm_Wi_f, 1024, 1, lstm_b_f, nullptr,0, xWf, 1024, B_*S_, 1024, E_, 0);
  gemm(stream, emb, E_, lstm_Wi_b, 1024, 1, lstm_b_b, nullptr,0, xWb, 1024, B_*S_, 1024, E_, 0);

  // ---- LSTM recurrences (z = h@Wh + xW_s via GEMM residual path) ----
  for (int s = 0; s < S_; ++s){
    gemm(stream, q, TWOH_, lstm_Wh_f, 1024, 1, nullptr, xWf + (size_t)s*1024, S_*1024,
         zbuf, 1024, B_, 1024, HL_, 0);
    k_lstm_gates<<<nblk(B_*HL_,TPB),TPB,0,stream>>>(zbuf, q_len, s, q, TWOH_, cstate,
                                                    lstmout, 0);
  }
  for (int s = S_-1; s >= 0; --s){
    gemm(stream, q + HL_, TWOH_, lstm_Wh_b, 1024, 1, nullptr, xWb + (size_t)s*1024, S_*1024,
         zbuf, 1024, B_, 1024, HL_, 0);
    k_lstm_gates<<<nblk(B_*HL_,TPB),TPB,0,stream>>>(zbuf, q_len, s, q + HL_, TWOH_,
                                                    cstate + B_*HL_, lstmout, HL_);
  }

  // ---- controller setup ----
  k_mixw<<<1, 32, 0, stream>>>(cell3_ow, cell4_ow, ans_ow, mixw);
  k_tile_c0<<<nblk(B_*TWOH_,TPB),TPB,0,stream>>>(c_0, ucat);

  // ---- T controller steps ----
  for (int t = 0; t < T_; ++t){
    gemm(stream, q, TWOH_, W1 + (size_t)t*TWOH_*TWOH_, TWOH_, 1, b_1, nullptr,0,
         ucat, 1024, B_, TWOH_, TWOH_, 0);
    gemm(stream, ucat, 1024, W2_w, TWOH_, 1, W2_b, nullptr,0,
         u, TWOH_, B_, TWOH_, 1024, 0);
    k_wl<<<B_, 32, 0, stream>>>(u, Wmod, wprob, ent);
    k_attn<<<B_, 256, 0, stream>>>(lstmout, u, W3, ucat);
    gemm(stream, ucat + 512, 1024, t2m_w, MAPD_, 1, t2m_b, nullptr,0,
         tmap, MAPD_, B_, MAPD_, TWOH_, 0);

    const float* pcur = (t & 1) ? pB : pA;
    float*       pnxt = (t & 1) ? pA : pB;

    k_atop<<<nblk(NPIX_,TPB),TPB,0,stream>>>(stack, pcur, atop, a2b);
    k_cellout<<<NPIX_, 256, 0, stream>>>(imgmap, tmap, atop, a2b, mixw,
                                         cell3_cw, cell3_cb, cell4_cw, cell4_cb, cellout);
    k_ansmix<<<nblk(B_*MAPD_,TPB),TPB,0,stream>>>(imgmap, tmap, atop, a2b, mixw,
                                                  mcat6, mcat7);
    gemm(stream, mcat6, 1024, ans_W,                      MEMD_, 1, ans_b,         nullptr,0,
         mem6, MEMD_, B_, MEMD_, 1024, 0);
    gemm(stream, mcat7, 1024, ans_W + (size_t)1024*MEMD_, MEMD_, 1, ans_b + MEMD_, nullptr,0,
         mem7, MEMD_, B_, MEMD_, 1024, 0);

    k_stack_update<<<nblk(NPIX_,TPB),TPB,0,stream>>>(stack, pcur, wprob, cellout);
    k_p_update<<<nblk(B_*L_,TPB),TPB,0,stream>>>(wprob, pcur, pnxt);
    k_mem_update<<<nblk(B_*MEMD_,TPB),TPB,0,stream>>>(wprob, mem6, mem7, mem);
  }

  // ---- output head ----
  k_qm<<<nblk((size_t)B_*1024,TPB),TPB,0,stream>>>(q, mem, qm);
  gemm(stream, qm, 1024, out1_w, MLP_, 1, out1_b, nullptr,0, h1, MLP_, B_, MLP_, 1024, 1);
  gemm(stream, h1, MLP_, out2_w, NCLS_, 1, out2_b, nullptr,0, out, NCLS_, B_, NCLS_, MLP_, 0);
  k_entred<<<1, 32, 0, stream>>>(ent, out + B_*NCLS_);
}
